// GlimpseNetwork_43370579755429
// MI455X (gfx1250) — compile-verified, matrix operand order checked by reading, not run on hardware
//
#include <hip/hip_runtime.h>
#include <hip/hip_bf16.h>

// Problem constants (from reference)
#define BB   128
#define CC   3
#define HH   224
#define WW   224
#define GG   32
#define D_IN 9216      // K*G*G*C = 3*32*32*3
#define HG   1024
#define HL   256
#define NCAT 1280      // HG + HL

typedef __attribute__((ext_vector_type(16))) __bf16 v16bf;
typedef __attribute__((ext_vector_type(8)))  __bf16 v8bf;
typedef __attribute__((ext_vector_type(8)))  float  v8f;
typedef __attribute__((ext_vector_type(4)))  float  v4f;

static __device__ __forceinline__ __bf16 f2bf(float f) {
    // round-to-nearest-even fp32 -> bf16
    union { float f; unsigned u; } a; a.f = f;
    unsigned r = a.u + 0x7FFFu + ((a.u >> 16) & 1u);
    unsigned short h = (unsigned short)(r >> 16);
    __bf16 o; __builtin_memcpy(&o, &h, 2);
    return o;
}

static __device__ __forceinline__ v16bf load_frag(const __bf16* p0, const __bf16* p1) {
    union { v16bf v; v8bf h[2]; } u;
    u.h[0] = *(const v8bf*)p0;   // 16B -> global_load_b128
    u.h[1] = *(const v8bf*)p1;
    return u.v;
}

// ---------------- weight conversion (8 elems/thread, b128 in / b128 out) ----------------
__global__ void conv_w1_kernel(const float* __restrict__ w1, __bf16* __restrict__ w1b, int n8) {
    int i = blockIdx.x * blockDim.x + threadIdx.x;   // n8 = n/8 groups
    if (i >= n8) return;
    v4f a = *(const v4f*)(w1 + (size_t)i * 8);
    v4f b = *(const v4f*)(w1 + (size_t)i * 8 + 4);
    v8bf o;
    o[0] = f2bf(a[0]); o[1] = f2bf(a[1]); o[2] = f2bf(a[2]); o[3] = f2bf(a[3]);
    o[4] = f2bf(b[0]); o[5] = f2bf(b[1]); o[6] = f2bf(b[2]); o[7] = f2bf(b[3]);
    *(v8bf*)(w1b + (size_t)i * 8) = o;
}

__global__ void conv_wcat_kernel(const float* __restrict__ w3, const float* __restrict__ w4,
                                 __bf16* __restrict__ wcat) {
    int i = blockIdx.x * blockDim.x + threadIdx.x;   // NCAT*NCAT/8 groups of 8
    if (i >= (NCAT * NCAT) / 8) return;
    int e    = i * 8;
    int nrow = e / NCAT;
    int k    = e - nrow * NCAT;                       // multiple of 8; 1024 boundary 8-aligned
    const float* src = (k < HG) ? (w3 + (size_t)nrow * HG + k)
                                : (w4 + (size_t)nrow * HL + (k - HG));
    v4f a = *(const v4f*)(src);
    v4f b = *(const v4f*)(src + 4);
    v8bf o;
    o[0] = f2bf(a[0]); o[1] = f2bf(a[1]); o[2] = f2bf(a[2]); o[3] = f2bf(a[3]);
    o[4] = f2bf(b[0]); o[5] = f2bf(b[1]); o[6] = f2bf(b[2]); o[7] = f2bf(b[3]);
    *(v8bf*)(wcat + (size_t)e) = o;
}

// ---------------- retina: patches -> bf16 phi ----------------
__global__ void patch_kernel(const float* __restrict__ x, const float* __restrict__ l,
                             __bf16* __restrict__ phi) {
    int idx = blockIdx.x * blockDim.x + threadIdx.x;   // BB * D_IN
    if (idx >= BB * D_IN) return;
    int b = idx / D_IN;
    int r = idx - b * D_IN;
    int k  = r / (CC * GG * GG);         // scale index 0..2
    int rem = r % (CC * GG * GG);
    int c  = rem / (GG * GG);
    int gy = (rem / GG) % GG;
    int gx = rem % GG;

    float lx = l[b * 2 + 0];
    float ly = l[b * 2 + 1];
    int sx = (int)(0.5f * ((lx + 1.0f) * (float)HH));
    int sy = (int)(0.5f * ((ly + 1.0f) * (float)HH));
    // dynamic_slice clamp: padded dim is H+2*pad, slice is `size` -> start in [0, H]
    sx = min(max(sx, 0), HH);
    sy = min(max(sy, 0), HH);

    int f   = 1 << k;        // pool factor 1,2,4
    int s   = GG << k;       // 32,64,128
    int pad = s >> 1;
    int row0 = sy - pad + gy * f;
    int col0 = sx - pad + gx * f;

    const float* xc = x + ((size_t)b * CC + c) * HH * WW;
    float sum = 0.0f;
    for (int dy = 0; dy < f; ++dy) {
        int rr = row0 + dy;
        if (rr < 0 || rr >= HH) continue;   // padded region = 0
        const float* xr = xc + (size_t)rr * WW;
        for (int dx = 0; dx < f; ++dx) {
            int cc = col0 + dx;
            if (cc >= 0 && cc < WW) sum += xr[cc];
        }
    }
    phi[idx] = f2bf(sum * (1.0f / (float)(f * f)));
}

// ---------------- union mask (nested rects -> largest), float4 stores ----------------
__global__ void mask_kernel(const float* __restrict__ l, float* __restrict__ mask) {
    int idx = blockIdx.x * blockDim.x + threadIdx.x;   // BB * HH * (WW/4)
    if (idx >= BB * HH * (WW / 4)) return;
    int b = idx / (HH * (WW / 4));
    int p = idx - b * (HH * (WW / 4));
    int rr = p / (WW / 4);
    int c0 = (p - rr * (WW / 4)) * 4;
    float lx = l[b * 2 + 0];
    float ly = l[b * 2 + 1];
    int sx = (int)(0.5f * ((lx + 1.0f) * (float)HH));
    int sy = (int)(0.5f * ((ly + 1.0f) * (float)HH));
    const int pad = 64;   // largest scale dominates the union
    bool iny = (rr >= sy - pad) && (rr < sy + pad);
    v4f o;
    #pragma unroll
    for (int j = 0; j < 4; ++j) {
        int cc = c0 + j;
        o[j] = (iny && cc >= sx - pad && cc < sx + pad) ? 1.0f : 0.0f;
    }
    *(v4f*)(mask + ((size_t)b * HH + rr) * WW + c0) = o;
}

// ---------------- tiny l-path: relu(l @ w2^T + b2) -> bf16 into acat[:,1024:] ----------------
__global__ void lout_kernel(const float* __restrict__ l, const float* __restrict__ w2,
                            const float* __restrict__ b2, __bf16* __restrict__ acat) {
    int idx = blockIdx.x * blockDim.x + threadIdx.x;   // BB * HL
    if (idx >= BB * HL) return;
    int b = idx / HL;
    int n = idx - b * HL;
    float v = l[b * 2 + 0] * w2[n * 2 + 0] + l[b * 2 + 1] * w2[n * 2 + 1] + b2[n];
    v = v > 0.0f ? v : 0.0f;
    acat[(size_t)b * NCAT + HG + n] = f2bf(v);
}

// ---------------- WMMA bf16 GEMM: out[m,n] = relu(sum_k A[m,k]*W[n,k] + bias) ----------------
// A: (M,K) bf16 row-major.  W: (N,K) bf16 row-major (weight layout == WMMA B columns).
// Each wave32 computes a 16(M) x 64(N) tile; double-buffered k-loop so the 10 fragment
// loads of step k+32 are in flight while the 4 WMMAs of step k execute.
__global__ void wmma_gemm_kernel(const __bf16* __restrict__ A, const __bf16* __restrict__ W,
                                 const float* __restrict__ bias0, const float* __restrict__ bias1,
                                 float* __restrict__ outF, __bf16* __restrict__ outB,
                                 int M, int N, int K, int ldo) {
    const int lane  = threadIdx.x & 31;
    const int wave  = threadIdx.x >> 5;
    const int wpb   = blockDim.x >> 5;
    const int tile  = blockIdx.x * wpb + wave;
    const int ntn   = N >> 6;               // 64-wide N tiles
    const int mt    = tile / ntn;
    const int nt    = tile - mt * ntn;
    if (mt * 16 >= M) return;

    const int half   = lane >> 4;           // which K-half this lane carries
    const int nlocal = lane & 15;           // row (A) / column (B) within 16
    const int k0off  = half * 8;            // frag elems 0..7  : K = half*8 + 0..7
    const int k1off  = 16 + half * 8;       // frag elems 8..15 : K = 16 + half*8 + 0..7

    const __bf16* arow  = A + (size_t)(mt * 16 + nlocal) * K;
    const __bf16* wrow0 = W + (size_t)(nt * 64 +  0 + nlocal) * K;
    const __bf16* wrow1 = W + (size_t)(nt * 64 + 16 + nlocal) * K;
    const __bf16* wrow2 = W + (size_t)(nt * 64 + 32 + nlocal) * K;
    const __bf16* wrow3 = W + (size_t)(nt * 64 + 48 + nlocal) * K;

    v8f acc0 = {}, acc1 = {}, acc2 = {}, acc3 = {};

    // prologue: load k=0 fragments
    v16bf af  = load_frag(arow  + k0off, arow  + k1off);
    v16bf bf0 = load_frag(wrow0 + k0off, wrow0 + k1off);
    v16bf bf1 = load_frag(wrow1 + k0off, wrow1 + k1off);
    v16bf bf2 = load_frag(wrow2 + k0off, wrow2 + k1off);
    v16bf bf3 = load_frag(wrow3 + k0off, wrow3 + k1off);

    for (int k = 32; k < K; k += 32) {
        // issue next-step loads before consuming current fragments
        v16bf afn = load_frag(arow  + k + k0off, arow  + k + k1off);
        v16bf bn0 = load_frag(wrow0 + k + k0off, wrow0 + k + k1off);
        v16bf bn1 = load_frag(wrow1 + k + k0off, wrow1 + k + k1off);
        v16bf bn2 = load_frag(wrow2 + k + k0off, wrow2 + k + k1off);
        v16bf bn3 = load_frag(wrow3 + k + k0off, wrow3 + k + k1off);

        acc0 = __builtin_amdgcn_wmma_f32_16x16x32_bf16(false, af, false, bf0, (short)0, acc0, false, false);
        acc1 = __builtin_amdgcn_wmma_f32_16x16x32_bf16(false, af, false, bf1, (short)0, acc1, false, false);
        acc2 = __builtin_amdgcn_wmma_f32_16x16x32_bf16(false, af, false, bf2, (short)0, acc2, false, false);
        acc3 = __builtin_amdgcn_wmma_f32_16x16x32_bf16(false, af, false, bf3, (short)0, acc3, false, false);

        af = afn; bf0 = bn0; bf1 = bn1; bf2 = bn2; bf3 = bn3;
    }
    // epilogue k-step
    acc0 = __builtin_amdgcn_wmma_f32_16x16x32_bf16(false, af, false, bf0, (short)0, acc0, false, false);
    acc1 = __builtin_amdgcn_wmma_f32_16x16x32_bf16(false, af, false, bf1, (short)0, acc1, false, false);
    acc2 = __builtin_amdgcn_wmma_f32_16x16x32_bf16(false, af, false, bf2, (short)0, acc2, false, false);
    acc3 = __builtin_amdgcn_wmma_f32_16x16x32_bf16(false, af, false, bf3, (short)0, acc3, false, false);

    // Epilogue: C/D layout -> lane has fixed n = nlocal, rows m = mt*16 + half*8 + r
    const int mbase = mt * 16 + half * 8;
    v8f accs[4] = { acc0, acc1, acc2, acc3 };
    #pragma unroll
    for (int j = 0; j < 4; ++j) {
        int n = nt * 64 + j * 16 + nlocal;
        float bs = bias0[n] + (bias1 ? bias1[n] : 0.0f);
        if (outF) {
            float* o = outF + (size_t)mbase * ldo + n;
            #pragma unroll
            for (int r = 0; r < 8; ++r) {
                float v = accs[j][r] + bs;
                o[(size_t)r * ldo] = v > 0.0f ? v : 0.0f;
            }
        } else {
            __bf16* o = outB + (size_t)mbase * ldo + n;
            #pragma unroll
            for (int r = 0; r < 8; ++r) {
                float v = accs[j][r] + bs;
                o[(size_t)r * ldo] = f2bf(v > 0.0f ? v : 0.0f);
            }
        }
    }
}

extern "C" void kernel_launch(void* const* d_in, const int* in_sizes, int n_in,
                              void* d_out, int out_size, void* d_ws, size_t ws_size,
                              hipStream_t stream) {
    const float* x  = (const float*)d_in[0];
    const float* l  = (const float*)d_in[1];
    const float* w1 = (const float*)d_in[2];
    const float* b1 = (const float*)d_in[3];
    const float* w2 = (const float*)d_in[4];
    const float* b2 = (const float*)d_in[5];
    const float* w3 = (const float*)d_in[6];
    const float* b3 = (const float*)d_in[7];
    const float* w4 = (const float*)d_in[8];
    const float* b4 = (const float*)d_in[9];

    float* g_t  = (float*)d_out;                     // (128, 1280)
    float* mask = (float*)d_out + (size_t)BB * NCAT; // (128, 224*224)

    // workspace carve-out (256B aligned)
    char*  ws  = (char*)d_ws;
    size_t off = 0;
    auto carve = [&](size_t bytes) -> void* {
        void* p = ws + off;
        off = (off + bytes + 255) & ~(size_t)255;
        return p;
    };
    __bf16* phi  = (__bf16*)carve((size_t)BB * D_IN * 2);      // 2.36 MB
    __bf16* w1b  = (__bf16*)carve((size_t)HG * D_IN * 2);      // 18.9 MB
    __bf16* wcat = (__bf16*)carve((size_t)NCAT * NCAT * 2);    // 3.28 MB
    __bf16* acat = (__bf16*)carve((size_t)BB * NCAT * 2);      // 0.33 MB

    const int T = 256;

    // 1) weights -> bf16 (vectorized, 8 elems/thread)
    {
        int n8 = (HG * D_IN) / 8;
        conv_w1_kernel<<<(n8 + T - 1) / T, T, 0, stream>>>(w1, w1b, n8);
    }
    conv_wcat_kernel<<<((NCAT * NCAT) / 8 + T - 1) / T, T, 0, stream>>>(w3, w4, wcat);

    // 2) retina patches -> bf16 phi
    patch_kernel<<<(BB * D_IN + T - 1) / T, T, 0, stream>>>(x, l, phi);

    // 3) mask directly to output (float4 stores)
    mask_kernel<<<(BB * HH * (WW / 4) + T - 1) / T, T, 0, stream>>>(l, mask);

    // 4) l-path into acat[:, 1024:1280]
    lout_kernel<<<(BB * HL + T - 1) / T, T, 0, stream>>>(l, w2, b2, acat);

    // 5) GEMM1: acat[:, :1024] = relu(phi @ w1^T + b1)   (M=128,N=1024,K=9216)
    {
        int tiles = (BB / 16) * (HG / 64);   // 128 wave-tiles
        wmma_gemm_kernel<<<tiles / 8, 256, 0, stream>>>(
            phi, w1b, b1, nullptr, nullptr, acat, BB, HG, D_IN, NCAT);
    }

    // 6) GEMM2: g_t = relu(acat @ wcat^T + b3 + b4)      (M=128,N=1280,K=1280)
    {
        int tiles = (BB / 16) * (NCAT / 64); // 160 wave-tiles
        wmma_gemm_kernel<<<tiles / 8, 256, 0, stream>>>(
            acat, wcat, b3, b4, g_t, nullptr, BB, NCAT, NCAT, NCAT);
    }
}
